// Experts_18863496364575
// MI455X (gfx1250) — matrix operand order repaired and checked by reading, not compiled, not run on hardware
//
#include <hip/hip_runtime.h>

// ---------------------------------------------------------------------------
// Experts FFN: out[b,e,n,:] = gelu(x[b,e,n,:] @ w1[e] + b1[e]) @ w2[e] + b2[e]
// B=4, E=8, N=1024, D=1024, H=4096; fp32 in/out, bf16 WMMA compute.
//
// Pipeline (all on `stream`):
//   1) transpose_cvt : w1 (E,D,H) f32 -> w1t (E,H,D) bf16     [ws +64 MiB]
//   2) transpose_cvt : w2 (E,H,D) f32 -> w2t (E,D,H) bf16     [ws +64 MiB]
//   3) xcvt          : x (B,E,N,D) f32 -> xb (E,B*N,D) bf16   [ws +64 MiB]
//   4) gemm_tile<1024,true>  : hid = gelu(xb @ w1t^T + b1)    [ws +256 MiB]
//   5) gemm_tile<4096,false> : out = hid @ w2t^T + b2  (fp32)
// 256x128 output tiles, LDS-staged 64-K chunks shared by 8 waves,
// 4x4 register tile per wave -> 1 ds_load_b128 per v_wmma.
// ---------------------------------------------------------------------------

static constexpr int Bb = 4;
static constexpr int Ee = 8;
static constexpr int Nn = 1024;
static constexpr int Dd = 1024;
static constexpr int Hh = 4096;
static constexpr int MROWS = Bb * Nn;       // 4096 token rows per expert
static constexpr int MT    = 256;           // block rows
static constexpr int NT    = 128;           // block cols
static constexpr int KC    = 64;            // K chunk staged per barrier
static constexpr int LPITCH = KC + 8;       // LDS row pitch (bf16): +16B pad

typedef __bf16 v16bf __attribute__((ext_vector_type(16)));
typedef float  v8f   __attribute__((ext_vector_type(8)));

union Frag {                 // 32 B = 16 bf16: one WMMA A/B operand per lane
    uint4 q[2];
    v16bf v;
};

__device__ __forceinline__ unsigned short f2bf(float f) {
    union { float f; unsigned u; } cv; cv.f = f;
    unsigned r = cv.u + 0x7FFFu + ((cv.u >> 16) & 1u);   // round-to-nearest-even
    return (unsigned short)(r >> 16);
}
__device__ __forceinline__ unsigned pack2bf(float lo, float hi) {
    return (unsigned)f2bf(lo) | ((unsigned)f2bf(hi) << 16);
}
__device__ __forceinline__ float gelu_tanh(float x) {
    float x3 = x * x * x;
    float t  = tanhf(0.7978845608028654f * (x + 0.044715f * x3));
    return 0.5f * x * (1.0f + t);
}

__device__ __forceinline__ void loadFragA(Frag& f, const unsigned short* p) {
    f.q[0] = *reinterpret_cast<const uint4*>(p);        // K + {0..7 | 8..15}
    f.q[1] = *reinterpret_cast<const uint4*>(p + 16);   // K + {16..23 | 24..31}
}
__device__ __forceinline__ void loadFragB(Frag& f, const unsigned short* p) {
    f.q[0] = *reinterpret_cast<const uint4*>(p);        // K + {0..7 | 8..15}
    f.q[1] = *reinterpret_cast<const uint4*>(p + 16);   // K + {16..23 | 24..31}
}
__device__ __forceinline__ v8f wmma_bf16(const Frag& a, const Frag& b, v8f c) {
    return __builtin_amdgcn_wmma_f32_16x16x32_bf16(false, a.v, false, b.v,
                                                   (short)0, c, false, false);
}

// ---------------------------------------------------------------------------
// Transpose + fp32->bf16: src (E,R,C) f32 -> dst (E,C,R) bf16. Block (32,8).
// ---------------------------------------------------------------------------
__global__ void transpose_cvt(const float* __restrict__ src,
                              unsigned short* __restrict__ dst, int R, int C) {
    __shared__ float tile[32][33];
    const int e  = blockIdx.z;
    const int c0 = blockIdx.x * 32;
    const int r0 = blockIdx.y * 32;
    const float*    s = src + (size_t)e * R * C;
    unsigned short* d = dst + (size_t)e * R * C;
    const int tx = threadIdx.x, ty = threadIdx.y;
#pragma unroll
    for (int i = 0; i < 32; i += 8)
        tile[ty + i][tx] = s[(size_t)(r0 + ty + i) * C + (c0 + tx)];
    __syncthreads();
#pragma unroll
    for (int i = 0; i < 32; i += 8)
        d[(size_t)(c0 + ty + i) * R + (r0 + tx)] = f2bf(tile[tx][ty + i]);
}

// ---------------------------------------------------------------------------
// x (B,E,N,D) f32 -> xb (E, B*N, D) bf16  (expert-major token rows)
// ---------------------------------------------------------------------------
__global__ void xcvt(const float* __restrict__ x, unsigned short* __restrict__ xb) {
    const size_t i = ((size_t)blockIdx.x * 256 + threadIdx.x) * 4;
    const int row = (int)(i >> 10);            // input token row (b,e,n)
    const int d   = (int)(i & (Dd - 1));
    const int b   = row >> 13;                 // E*N = 8192 rows per b
    const int e   = (row >> 10) & (Ee - 1);
    const int n   = row & (Nn - 1);
    float4 v = *reinterpret_cast<const float4*>(x + i);
    uint2 p;
    p.x = pack2bf(v.x, v.y);
    p.y = pack2bf(v.z, v.w);
    const size_t orow = (size_t)e * MROWS + (size_t)b * Nn + n;
    *reinterpret_cast<uint2*>(xb + orow * Dd + d) = p;
}

// ---------------------------------------------------------------------------
// 256x128 tile GEMM:  C = act(A @ Bw^T + bias)
//   A  : (E, 4096, KTOT) bf16, K contiguous
//   Bw : (E, NTOT, KTOT) bf16, K contiguous (pre-transposed weights)
//   GELU=true : C = hid (E,4096,NTOT) bf16, gelu epilogue
//   GELU=false: C = out (B,E,N,D) f32, bias epilogue
// blockIdx.x = nblk*16 + mblk  (m fastest -> adjacent blocks share B tile in L2)
// blockIdx.y = expert
// ---------------------------------------------------------------------------
template <int KTOT, bool GELU>
__global__ __launch_bounds__(256) void gemm_tile(
        const unsigned short* __restrict__ A,
        const unsigned short* __restrict__ Bw,
        const float* __restrict__ bias,
        void* __restrict__ Cout, int NTOT) {
    __shared__ unsigned short aT[MT * LPITCH];   // 36 KB
    __shared__ unsigned short bT[NT * LPITCH];   // 18 KB

    const int e    = blockIdx.y;
    const int mblk = blockIdx.x & 15;            // 4096/256 = 16 row blocks
    const int nblk = blockIdx.x >> 4;
    const int m0   = mblk * MT;
    const int n0   = nblk * NT;

    const unsigned short* Ae = A  + ((size_t)e * MROWS + m0) * KTOT;
    const unsigned short* Be = Bw + ((size_t)e * NTOT  + n0) * KTOT;

    const int t     = threadIdx.x;
    const int wave  = t >> 5;
    const int ln    = t & 31;
    const int lhalf = ln >> 4;        // which 16-lane half
    const int lcol  = ln & 15;
    const int wr    = wave >> 1;      // 4 row groups x 4 row tiles (64 rows)
    const int wc    = wave & 1;       // 2 col groups x 4 col tiles (64 cols)

    // staging: A = 1 thread per 256-row (64 bf16 = 128 B each)
    //          B = 2 threads per 128-row (32 bf16 =  64 B each)
    const int aRow = t;               // 0..255
    const int bRow = t >> 1;          // 0..127
    const int bOff = (t & 1) * 32;    // bf16 elements

    v8f acc[4][4];
#pragma unroll
    for (int rt = 0; rt < 4; ++rt)
#pragma unroll
        for (int ct = 0; ct < 4; ++ct)
            acc[rt][ct] = v8f{0, 0, 0, 0, 0, 0, 0, 0};

    for (int k0 = 0; k0 < KTOT; k0 += KC) {
        // ---- cooperative staging: pure b128 copies (bf16 -> bf16) ---------
        const unsigned short* ag = Ae + (size_t)aRow * KTOT + k0;
        const unsigned short* bg = Be + (size_t)bRow * KTOT + k0 + bOff;
        unsigned short* as = &aT[aRow * LPITCH];
        unsigned short* bs = &bT[bRow * LPITCH + bOff];
#pragma unroll
        for (int i = 0; i < 64; i += 8)
            *reinterpret_cast<uint4*>(as + i) = *reinterpret_cast<const uint4*>(ag + i);
#pragma unroll
        for (int i = 0; i < 32; i += 8)
            *reinterpret_cast<uint4*>(bs + i) = *reinterpret_cast<const uint4*>(bg + i);
        if (k0 + KC < KTOT) {                       // pull next chunk toward L2/L0
            __builtin_prefetch(ag + KC, 0, 0);
            __builtin_prefetch(bg + KC, 0, 0);
        }
        __syncthreads();

        // ---- 4x4 register tile per wave, two 32-K steps -------------------
#pragma unroll
        for (int kk = 0; kk < KC; kk += 32) {
            Frag af[4];
#pragma unroll
            for (int rt = 0; rt < 4; ++rt)
                loadFragA(af[rt], &aT[(wr * 64 + rt * 16 + lcol) * LPITCH + kk + lhalf * 8]);
#pragma unroll
            for (int ct = 0; ct < 4; ++ct) {
                Frag bf;
                loadFragB(bf, &bT[(wc * 64 + ct * 16 + lcol) * LPITCH + kk + lhalf * 8]);
#pragma unroll
                for (int rt = 0; rt < 4; ++rt)
                    acc[rt][ct] = wmma_bf16(af[rt], bf, acc[rt][ct]);
            }
        }
        __syncthreads();
    }

    // ---- epilogue ---------------------------------------------------------
    if (GELU) {
        unsigned short* hout = reinterpret_cast<unsigned short*>(Cout)
                               + ((size_t)e * MROWS + m0) * NTOT + n0;
#pragma unroll
        for (int ct = 0; ct < 4; ++ct) {
            const int   col = wc * 64 + ct * 16 + lcol;
            const float bb  = bias[(size_t)e * NTOT + n0 + col];
#pragma unroll
            for (int rt = 0; rt < 4; ++rt)
#pragma unroll
                for (int r = 0; r < 8; ++r) {
                    const int m = wr * 64 + rt * 16 + r + lhalf * 8;   // C layout
                    hout[(size_t)m * NTOT + col] = f2bf(gelu_tanh(acc[rt][ct][r] + bb));
                }
        }
    } else {
        // token row m0+m -> (b = row/1024, n = row%1024); 256 | 1024 so b const
        const int bq = m0 >> 10;
        const int nq = m0 & (Nn - 1);
        float* obase = reinterpret_cast<float*>(Cout)
                       + (((size_t)bq * Ee + e) * Nn + nq) * Dd + n0;
#pragma unroll
        for (int ct = 0; ct < 4; ++ct) {
            const int   col = wc * 64 + ct * 16 + lcol;
            const float bb  = bias[(size_t)e * NTOT + n0 + col];
#pragma unroll
            for (int rt = 0; rt < 4; ++rt)
#pragma unroll
                for (int r = 0; r < 8; ++r) {
                    const int m = wr * 64 + rt * 16 + r + lhalf * 8;
                    obase[(size_t)m * Dd + col] = acc[rt][ct][r] + bb;
                }
        }
    }
}

// ---------------------------------------------------------------------------
extern "C" void kernel_launch(void* const* d_in, const int* in_sizes, int n_in,
                              void* d_out, int out_size, void* d_ws, size_t ws_size,
                              hipStream_t stream) {
    const float* x  = (const float*)d_in[0];   // (B,E,N,D)
    const float* w1 = (const float*)d_in[1];   // (E,D,H)
    const float* b1 = (const float*)d_in[2];   // (E,H)
    const float* w2 = (const float*)d_in[3];   // (E,H,D)
    const float* b2 = (const float*)d_in[4];   // (E,D)
    float* out = (float*)d_out;

    // workspace carve-up (448 MiB total)
    unsigned short* w1t = (unsigned short*)d_ws;                 //  64 MiB (E,H,D)
    unsigned short* w2t = w1t + (size_t)Ee * Hh * Dd;            //  64 MiB (E,D,H)
    unsigned short* xb  = w2t + (size_t)Ee * Dd * Hh;            //  64 MiB (E,BN,D)
    unsigned short* hid = xb  + (size_t)Ee * MROWS * Dd;         // 256 MiB (E,BN,H)

    dim3 tb(32, 8);
    transpose_cvt<<<dim3(Hh / 32, Dd / 32, Ee), tb, 0, stream>>>(w1, w1t, Dd, Hh);
    transpose_cvt<<<dim3(Dd / 32, Hh / 32, Ee), tb, 0, stream>>>(w2, w2t, Hh, Dd);

    const int xblocks = (Bb * Ee * Nn * Dd) / (4 * 256);
    xcvt<<<xblocks, 256, 0, stream>>>(x, xb);

    // hid = gelu(xb @ w1t^T + b1) : K=1024, N=4096
    gemm_tile<Dd, true><<<dim3((MROWS / MT) * (Hh / NT), Ee), 256, 0, stream>>>(
        xb, w1t, b1, (void*)hid, Hh);

    // out = hid @ w2t^T + b2 : K=4096, N=1024
    gemm_tile<Hh, false><<<dim3((MROWS / MT) * (Dd / NT), Ee), 256, 0, stream>>>(
        hid, w2t, b2, (void*)out, Dd);
}
